// Summarizer_45389214384681
// MI455X (gfx1250) — matrix operand order repaired and checked by reading, not compiled, not run on hardware
//
#include <hip/hip_runtime.h>

#define SEQ  2048
#define DD   1024
#define HH   1024
#define GG   4096
#define NBLK 64

typedef __attribute__((ext_vector_type(16))) __bf16 v16bf;
typedef __attribute__((ext_vector_type(8)))  float  v8f;

union BFrag { uint4 q[2]; v16bf v; };

__device__ inline float bf2f(unsigned int v) {
    union { unsigned int i; float f; } u; u.i = v << 16; return u.f;
}
__device__ inline unsigned short f2bf(float f) {
    union { float f; unsigned int i; } u; u.f = f;
    unsigned int r = u.i + 0x7fffu + ((u.i >> 16) & 1u);
    return (unsigned short)(r >> 16);
}
__device__ inline float sigm(float x) { return 1.f / (1.f + __expf(-x)); }

// device-wide barrier (sense via monotonic generation counter)
__device__ inline void grid_barrier(int* bar, int nb) {
    __syncthreads();
    __threadfence();
    if (threadIdx.x == 0) {
        int g = __hip_atomic_load(&bar[1], __ATOMIC_ACQUIRE, __HIP_MEMORY_SCOPE_AGENT);
        if (atomicAdd(&bar[0], 1) == nb - 1) {
            __hip_atomic_store(&bar[0], 0, __ATOMIC_RELAXED, __HIP_MEMORY_SCOPE_AGENT);
            __hip_atomic_fetch_add(&bar[1], 1, __ATOMIC_RELEASE, __HIP_MEMORY_SCOPE_AGENT);
        } else {
            while (__hip_atomic_load(&bar[1], __ATOMIC_ACQUIRE, __HIP_MEMORY_SCOPE_AGENT) == g)
                __builtin_amdgcn_s_sleep(1);
        }
    }
    __syncthreads();
}

// dot of 8*N4 bf16 weights against fp32 vector
template<int N4>
__device__ inline float dotbf(const unsigned short* __restrict__ w, const float* __restrict__ v) {
    const uint4* wp = (const uint4*)w;
    float acc = 0.f;
#pragma unroll 8
    for (int i = 0; i < N4; ++i) {
        uint4 q = wp[i];
        const float* hv = v + i * 8;
        acc += bf2f(q.x) * hv[0] + bf2f(q.x >> 16) * hv[1]
             + bf2f(q.y) * hv[2] + bf2f(q.y >> 16) * hv[3]
             + bf2f(q.z) * hv[4] + bf2f(q.z >> 16) * hv[5]
             + bf2f(q.w) * hv[6] + bf2f(q.w >> 16) * hv[7];
    }
    return acc;
}

// ---------------- fp32 -> bf16 convert ----------------
__global__ __launch_bounds__(256) void k_f2bf(const float* __restrict__ s,
                                              unsigned short* __restrict__ d, int n) {
    int i = blockIdx.x * 256 + threadIdx.x;
    if (i < n) d[i] = f2bf(s[i]);
}

// ---------------- bf16 WMMA GEMM: C[M,N] = A[M,K] @ B[N,K]^T + bias1 + bias2 ----------
// block = 128 threads (4 waves); wave computes a 16x64 tile; grid = (M/16, N/256)
// Software-pipelined: fragments for iteration k+1 are loaded while the four
// WMMAs of iteration k execute, so v_wmma does not stall on s_wait_loadcnt 0.
__global__ __launch_bounds__(128) void k_gemm_bf16(
    const unsigned short* __restrict__ A, const unsigned short* __restrict__ B,
    const float* __restrict__ bias1, const float* __restrict__ bias2,
    float* __restrict__ C, int N, int K)
{
    const int lane = threadIdx.x & 31;
    const int wave = threadIdx.x >> 5;
    const int m0 = blockIdx.x * 16;
    const int n0 = blockIdx.y * 256 + wave * 64;

    v8f acc[4] = {};
    // A fragment addressing (16x32 bf16, ISA 7.12.2)
    const int arow = m0 + (lane & 15);
    const int asel = (lane >> 4) * 8;          // lanes>=16 start at K=8 within 32-chunk
    const unsigned short* ap = A + (size_t)arow * K + asel;
    // B fragment addressing (32x16 bf16): lane<16 -> K 0..7,16..23; lane>=16 -> K 8..15,24..31
    const unsigned short* bp0 = B + (size_t)(n0 + (lane & 15)) * K + ((lane >> 4) * 8);

    auto loadA = [&](int kb, BFrag& f) {
        f.q[0] = *(const uint4*)(ap + kb);
        f.q[1] = *(const uint4*)(ap + kb + 16);
    };
    auto loadB = [&](int kb, int j, BFrag& f) {
        const unsigned short* bp = bp0 + (size_t)j * 16 * K + kb;
        f.q[0] = ((const uint4*)bp)[0];
        f.q[1] = ((const uint4*)bp)[2];
    };

    BFrag fa, fb0, fb1, fb2, fb3;
    loadA(0, fa);
    loadB(0, 0, fb0); loadB(0, 1, fb1); loadB(0, 2, fb2); loadB(0, 3, fb3);

    for (int kb = 32; kb < K; kb += 32) {
        BFrag na, nb0, nb1, nb2, nb3;
        loadA(kb, na);
        loadB(kb, 0, nb0); loadB(kb, 1, nb1); loadB(kb, 2, nb2); loadB(kb, 3, nb3);
        acc[0] = __builtin_amdgcn_wmma_f32_16x16x32_bf16(false, fa.v, false, fb0.v, (short)0, acc[0], false, false);
        acc[1] = __builtin_amdgcn_wmma_f32_16x16x32_bf16(false, fa.v, false, fb1.v, (short)0, acc[1], false, false);
        acc[2] = __builtin_amdgcn_wmma_f32_16x16x32_bf16(false, fa.v, false, fb2.v, (short)0, acc[2], false, false);
        acc[3] = __builtin_amdgcn_wmma_f32_16x16x32_bf16(false, fa.v, false, fb3.v, (short)0, acc[3], false, false);
        fa = na; fb0 = nb0; fb1 = nb1; fb2 = nb2; fb3 = nb3;
    }
    acc[0] = __builtin_amdgcn_wmma_f32_16x16x32_bf16(false, fa.v, false, fb0.v, (short)0, acc[0], false, false);
    acc[1] = __builtin_amdgcn_wmma_f32_16x16x32_bf16(false, fa.v, false, fb1.v, (short)0, acc[1], false, false);
    acc[2] = __builtin_amdgcn_wmma_f32_16x16x32_bf16(false, fa.v, false, fb2.v, (short)0, acc[2], false, false);
    acc[3] = __builtin_amdgcn_wmma_f32_16x16x32_bf16(false, fa.v, false, fb3.v, (short)0, acc[3], false, false);

    const int crow = m0 + (lane >> 4) * 8;
#pragma unroll
    for (int j = 0; j < 4; ++j) {
        const int col = n0 + j * 16 + (lane & 15);
        const float bb = bias1[col] + bias2[col];
#pragma unroll
        for (int r = 0; r < 8; ++r)
            C[(size_t)(crow + r) * N + col] = acc[j][r] + bb;
    }
}

// ---------------- persistent LSTM scan ----------------
// 64 blocks x 256 thr. Block owns 16 hidden units (64 gate rows); 4 threads per row dot.
__global__ __launch_bounds__(256) void k_lstm_scan(
    const float* __restrict__ xg, const unsigned short* __restrict__ whh,
    float* __restrict__ h_buf,                 // 2048 floats (double buffered)
    float* __restrict__ out_f32, unsigned short* __restrict__ out_bf16,
    int colOff, int outStride, int reverse,
    float* __restrict__ h_last, float* __restrict__ c_last, int* bar)
{
    __shared__ float h_s[1024];
    __shared__ float dots[64];
    const int tid = threadIdx.x, blk = blockIdx.x, nb = gridDim.x;
    const int d = tid >> 2, q = tid & 3;
    const int u_l = d >> 2, g = d & 3;
    const int row = g * 1024 + blk * 16 + u_l;
    const unsigned short* wrow = whh + (size_t)row * 1024 + q * 256;
    float c = 0.f, hreg = 0.f;

    if (tid < 16) h_buf[blk * 16 + tid] = 0.f;
    grid_barrier(bar, nb);

    for (int t = 0; t < SEQ; ++t) {
        const int trow = reverse ? (SEQ - 1 - t) : t;
        const int rd = (t & 1) * 1024, wr = rd ^ 1024;
        for (int i = tid; i < 1024; i += 256) h_s[i] = h_buf[rd + i];
        // prefetch next step's xg row while this step computes
        if (t + 1 < SEQ) {
            const int nrow = reverse ? (SEQ - 2 - t) : (t + 1);
            __builtin_prefetch(xg + (size_t)nrow * GG + tid * 16, 0, 1);
        }
        __syncthreads();
        float acc = dotbf<32>(wrow, h_s + q * 256);
        acc += __shfl_xor(acc, 1, 32);
        acc += __shfl_xor(acc, 2, 32);
        if (q == 0) dots[d] = acc;
        __syncthreads();
        if (tid < 16) {
            const int unit = blk * 16 + tid;
            const float* xr = xg + (size_t)trow * GG;
            float gi = sigm (xr[unit]        + dots[tid * 4 + 0]);
            float gf = sigm (xr[1024 + unit] + dots[tid * 4 + 1]);
            float gc = tanhf(xr[2048 + unit] + dots[tid * 4 + 2]);
            float go = sigm (xr[3072 + unit] + dots[tid * 4 + 3]);
            c = gf * c + gi * gc;
            hreg = go * tanhf(c);
            h_buf[wr + unit] = hreg;
            size_t oidx = (size_t)trow * outStride + colOff + unit;
            if (out_f32)  out_f32[oidx]  = hreg;
            if (out_bf16) out_bf16[oidx] = f2bf(hreg);
        }
        grid_barrier(bar, nb);
    }
    if (tid < 16) {
        const int unit = blk * 16 + tid;
        if (h_last) h_last[unit] = hreg;
        if (c_last) c_last[unit] = c;
    }
}

// ---------------- scores = sigmoid(y1 @ soW + b) ----------------
__global__ __launch_bounds__(256) void k_scores(const float* __restrict__ y,
    const float* __restrict__ W, const float* __restrict__ b,
    float* __restrict__ out1, float* __restrict__ out2)
{
    __shared__ float red[256];
    const int s = blockIdx.x, tid = threadIdx.x;
    const float* yr = y + (size_t)s * 2048;
    float acc = 0.f;
    for (int k = tid; k < 2048; k += 256) acc += yr[k] * W[k];
    red[tid] = acc; __syncthreads();
    for (int w = 128; w > 0; w >>= 1) { if (tid < w) red[tid] += red[tid + w]; __syncthreads(); }
    if (tid == 0) {
        float v = sigm(red[0] + b[0]);
        out1[s] = v; out2[s] = v;
    }
}

__global__ __launch_bounds__(256) void k_weighted(const float* __restrict__ x,
    const float* __restrict__ sc, unsigned short* __restrict__ o, int n)
{
    int i = blockIdx.x * 256 + threadIdx.x;
    if (i < n) o[i] = f2bf(x[i] * sc[i >> 10]);
}

// ---------------- mu / log_var / z ----------------
__global__ __launch_bounds__(256) void k_muvar(
    const float* __restrict__ h0l, const float* __restrict__ h1l,
    const float* __restrict__ muW, const float* __restrict__ mub,
    const float* __restrict__ varW, const float* __restrict__ varb,
    const float* __restrict__ eps,
    float* __restrict__ mu_out, float* __restrict__ lv_out, float* __restrict__ z)
{
    int gt = blockIdx.x * 256 + threadIdx.x;   // 0..2047
    int r = gt >> 10, j = gt & 1023;
    const float* h  = r ? h1l : h0l;
    const float* wm = muW  + (size_t)j * HH;
    const float* wv = varW + (size_t)j * HH;
    float am = 0.f, av = 0.f;
    for (int k = 0; k < HH; ++k) { float hv = h[k]; am += wm[k] * hv; av += wv[k] * hv; }
    am += mub[j]; av += varb[j];
    mu_out[gt] = am; lv_out[gt] = av;
    z[gt] = am + eps[gt] * __expf(0.5f * av);
}

// ---------------- persistent 2-layer decoder ----------------
__global__ __launch_bounds__(256) void k_decoder(
    const unsigned short* __restrict__ wih0, const unsigned short* __restrict__ whh0,
    const unsigned short* __restrict__ wih1, const unsigned short* __restrict__ whh1,
    const unsigned short* __restrict__ wout,
    const float* __restrict__ dbih, const float* __restrict__ dbhh,
    const float* __restrict__ outb,
    const float* __restrict__ z, const float* __restrict__ c0init, const float* __restrict__ c1init,
    float* __restrict__ h0_buf, float* __restrict__ h1_buf, float* __restrict__ xt_buf,
    float* __restrict__ decoded, int* bar)
{
    __shared__ float a_s[1024];
    __shared__ float b_s[1024];
    __shared__ float dots[64];
    const int tid = threadIdx.x, blk = blockIdx.x, nb = gridDim.x;
    const int d = tid >> 2, q = tid & 3;
    const int u_l = d >> 2, g = d & 3;
    const int row = g * 1024 + blk * 16 + u_l;
    const int mat = q >> 1, half = q & 1;
    float c0 = 0.f, c1 = 0.f;

    if (tid < 16) {
        int unit = blk * 16 + tid;
        c0 = c0init[unit]; c1 = c1init[unit];
        h0_buf[unit] = z[unit];
        h1_buf[unit] = z[HH + unit];
    }
    if (blk == 0) for (int i = tid; i < DD; i += 256) xt_buf[i] = 0.f;
    grid_barrier(bar, nb);

    for (int k = 0; k < SEQ; ++k) {
        const int rd = (k & 1) * 1024, wr = rd ^ 1024;
        // ---- layer 0: gates = Wih0@xt + Whh0@h0 + b ----
        for (int i = tid; i < 1024; i += 256) { a_s[i] = xt_buf[i]; b_s[i] = h0_buf[rd + i]; }
        __syncthreads();
        {
            const unsigned short* w = (mat ? whh0 : wih0) + (size_t)row * 1024 + half * 512;
            const float* v = (mat ? b_s : a_s) + half * 512;
            float acc = dotbf<64>(w, v);
            acc += __shfl_xor(acc, 1, 32);
            acc += __shfl_xor(acc, 2, 32);
            if (q == 0) dots[d] = acc;
        }
        __syncthreads();
        if (tid < 16) {
            int unit = blk * 16 + tid;
            int r0 = blk * 16 + tid;           // layer 0 bias base
            float gi = sigm (dots[tid*4+0] + dbih[r0]        + dbhh[r0]);
            float gf = sigm (dots[tid*4+1] + dbih[r0 + 1024] + dbhh[r0 + 1024]);
            float gc = tanhf(dots[tid*4+2] + dbih[r0 + 2048] + dbhh[r0 + 2048]);
            float go = sigm (dots[tid*4+3] + dbih[r0 + 3072] + dbhh[r0 + 3072]);
            c0 = gf * c0 + gi * gc;
            h0_buf[wr + unit] = go * tanhf(c0);
        }
        grid_barrier(bar, nb);
        // ---- layer 1: gates = Wih1@h0n + Whh1@h1 + b ----
        for (int i = tid; i < 1024; i += 256) { a_s[i] = h0_buf[wr + i]; b_s[i] = h1_buf[rd + i]; }
        __syncthreads();
        {
            const unsigned short* w = (mat ? whh1 : wih1) + (size_t)row * 1024 + half * 512;
            const float* v = (mat ? b_s : a_s) + half * 512;
            float acc = dotbf<64>(w, v);
            acc += __shfl_xor(acc, 1, 32);
            acc += __shfl_xor(acc, 2, 32);
            if (q == 0) dots[d] = acc;
        }
        __syncthreads();
        if (tid < 16) {
            int unit = blk * 16 + tid;
            int r0 = GG + blk * 16 + tid;      // layer 1 bias base
            float gi = sigm (dots[tid*4+0] + dbih[r0]        + dbhh[r0]);
            float gf = sigm (dots[tid*4+1] + dbih[r0 + 1024] + dbhh[r0 + 1024]);
            float gc = tanhf(dots[tid*4+2] + dbih[r0 + 2048] + dbhh[r0 + 2048]);
            float go = sigm (dots[tid*4+3] + dbih[r0 + 3072] + dbhh[r0 + 3072]);
            c1 = gf * c1 + gi * gc;
            h1_buf[wr + unit] = go * tanhf(c1);
        }
        grid_barrier(bar, nb);
        // ---- output projection: xn = Wout @ h1n + b ----
        for (int i = tid; i < 1024; i += 256) a_s[i] = h1_buf[wr + i];
        __syncthreads();
        {
            const int orow = blk * 16 + (tid >> 4), seg = tid & 15;
            float acc = dotbf<8>(wout + (size_t)orow * 1024 + seg * 64, a_s + seg * 64);
            acc += __shfl_xor(acc, 1, 32);
            acc += __shfl_xor(acc, 2, 32);
            acc += __shfl_xor(acc, 4, 32);
            acc += __shfl_xor(acc, 8, 32);
            if (seg == 0) {
                float xn = acc + outb[orow];
                decoded[(size_t)(SEQ - 1 - k) * DD + orow] = xn;
                xt_buf[orow] = xn;
            }
        }
        grid_barrier(bar, nb);
    }
}

// =================================================================
extern "C" void kernel_launch(void* const* d_in, const int* in_sizes, int n_in,
                              void* d_out, int out_size, void* d_ws, size_t ws_size,
                              hipStream_t stream) {
    (void)in_sizes; (void)n_in; (void)out_size; (void)ws_size;
    const float* x     = (const float*)d_in[0];
    const float* eps   = (const float*)d_in[1];
    const float* s0Wih = (const float*)d_in[2];
    const float* s0Whh = (const float*)d_in[3];
    const float* s0bih = (const float*)d_in[4];
    const float* s0bhh = (const float*)d_in[5];
    const float* s1Wih = (const float*)d_in[6];
    const float* s1Whh = (const float*)d_in[7];
    const float* s1bih = (const float*)d_in[8];
    const float* s1bhh = (const float*)d_in[9];
    const float* soW   = (const float*)d_in[10];
    const float* sob   = (const float*)d_in[11];
    const float* eWih  = (const float*)d_in[12];
    const float* eWhh  = (const float*)d_in[13];
    const float* ebih  = (const float*)d_in[14];
    const float* ebhh  = (const float*)d_in[15];
    const float* muW   = (const float*)d_in[16];
    const float* mub   = (const float*)d_in[17];
    const float* varW  = (const float*)d_in[18];
    const float* varb  = (const float*)d_in[19];
    const float* dWih  = (const float*)d_in[20];
    const float* dWhh  = (const float*)d_in[21];
    const float* dbih  = (const float*)d_in[22];
    const float* dbhh  = (const float*)d_in[23];
    const float* doW   = (const float*)d_in[24];
    const float* dob   = (const float*)d_in[25];

    size_t off = 0;
    auto alloc = [&](size_t bytes) -> void* {
        off = (off + 255) & ~(size_t)255;
        void* p = (char*)d_ws + off; off += bytes; return p;
    };
    int*            bar     = (int*)           alloc(256);
    unsigned short* x_bf    = (unsigned short*)alloc((size_t)SEQ * DD * 2);
    unsigned short* wbuf    = (unsigned short*)alloc((size_t)GG * 2048 * 2);
    unsigned short* whhb    = (unsigned short*)alloc((size_t)GG * HH * 2);
    unsigned short* dwih0b  = (unsigned short*)alloc((size_t)GG * HH * 2);
    unsigned short* dwih1b  = (unsigned short*)alloc((size_t)GG * HH * 2);
    unsigned short* dwhh0b  = (unsigned short*)alloc((size_t)GG * HH * 2);
    unsigned short* dwhh1b  = (unsigned short*)alloc((size_t)GG * HH * 2);
    unsigned short* doWb    = (unsigned short*)alloc((size_t)DD * HH * 2);
    float*          xg      = (float*)         alloc((size_t)SEQ * GG * 4);
    unsigned short* y0_bf   = (unsigned short*)alloc((size_t)SEQ * 2 * HH * 2);
    float*          y1      = (float*)         alloc((size_t)SEQ * 2 * HH * 4);
    unsigned short* wtd_bf  = (unsigned short*)alloc((size_t)SEQ * DD * 2);
    unsigned short* y0e_bf  = (unsigned short*)alloc((size_t)SEQ * HH * 2);
    float*          sc_ws   = (float*)alloc((size_t)SEQ * 4);
    float*          h_scan  = (float*)alloc(2048 * 4);
    float*          h0d     = (float*)alloc(2048 * 4);
    float*          h1d     = (float*)alloc(2048 * 4);
    float*          xt      = (float*)alloc(1024 * 4);
    float*          h0l     = (float*)alloc(1024 * 4);
    float*          c0l     = (float*)alloc(1024 * 4);
    float*          h1l     = (float*)alloc(1024 * 4);
    float*          c1l     = (float*)alloc(1024 * 4);
    float*          zbuf    = (float*)alloc(2048 * 4);

    float* out        = (float*)d_out;
    float* scores_out = out;            // (S,1)
    float* mu_out     = out + 2048;     // (2,H)
    float* lv_out     = out + 4096;     // (2,H)
    float* dec_out    = out + 6144;     // (S,1,D)

    hipMemsetAsync(bar, 0, 256, stream);

    auto conv = [&](const float* s, unsigned short* dst, size_t n) {
        k_f2bf<<<dim3((unsigned)((n + 255) / 256)), dim3(256), 0, stream>>>(s, dst, (int)n);
    };

    conv(x, x_bf, (size_t)SEQ * DD);

    // ---- stage 0 bi-LSTM ----
    for (int dir = 0; dir < 2; ++dir) {
        conv(s0Wih + (size_t)dir * GG * DD, wbuf, (size_t)GG * DD);
        k_gemm_bf16<<<dim3(SEQ / 16, GG / 256), 128, 0, stream>>>(
            x_bf, wbuf, s0bih + dir * GG, s0bhh + dir * GG, xg, GG, DD);
        conv(s0Whh + (size_t)dir * GG * HH, whhb, (size_t)GG * HH);
        k_lstm_scan<<<NBLK, 256, 0, stream>>>(xg, whhb, h_scan,
            nullptr, y0_bf, dir * HH, 2 * HH, dir, nullptr, nullptr, bar);
    }
    // ---- stage 1 bi-LSTM ----
    for (int dir = 0; dir < 2; ++dir) {
        conv(s1Wih + (size_t)dir * GG * 2 * HH, wbuf, (size_t)GG * 2 * HH);
        k_gemm_bf16<<<dim3(SEQ / 16, GG / 256), 128, 0, stream>>>(
            y0_bf, wbuf, s1bih + dir * GG, s1bhh + dir * GG, xg, GG, 2 * HH);
        conv(s1Whh + (size_t)dir * GG * HH, whhb, (size_t)GG * HH);
        k_lstm_scan<<<NBLK, 256, 0, stream>>>(xg, whhb, h_scan,
            y1, nullptr, dir * HH, 2 * HH, dir, nullptr, nullptr, bar);
    }
    // ---- attention scores + weighting ----
    k_scores<<<SEQ, 256, 0, stream>>>(y1, soW, sob, scores_out, sc_ws);
    k_weighted<<<(SEQ * DD) / 256, 256, 0, stream>>>(x, sc_ws, wtd_bf, SEQ * DD);
    // ---- encoder layer 0 ----
    conv(eWih, wbuf, (size_t)GG * DD);
    k_gemm_bf16<<<dim3(SEQ / 16, GG / 256), 128, 0, stream>>>(
        wtd_bf, wbuf, ebih, ebhh, xg, GG, DD);
    conv(eWhh, whhb, (size_t)GG * HH);
    k_lstm_scan<<<NBLK, 256, 0, stream>>>(xg, whhb, h_scan,
        nullptr, y0e_bf, 0, HH, 0, h0l, c0l, bar);
    // ---- encoder layer 1 ----
    conv(eWih + (size_t)GG * DD, wbuf, (size_t)GG * HH);
    k_gemm_bf16<<<dim3(SEQ / 16, GG / 256), 128, 0, stream>>>(
        y0e_bf, wbuf, ebih + GG, ebhh + GG, xg, GG, HH);
    conv(eWhh + (size_t)GG * HH, whhb, (size_t)GG * HH);
    k_lstm_scan<<<NBLK, 256, 0, stream>>>(xg, whhb, h_scan,
        nullptr, nullptr, 0, HH, 0, h1l, c1l, bar);
    // ---- latent ----
    k_muvar<<<8, 256, 0, stream>>>(h0l, h1l, muW, mub, varW, varb, eps,
                                   mu_out, lv_out, zbuf);
    // ---- decoder ----
    conv(dWih, dwih0b, (size_t)GG * HH);
    conv(dWih + (size_t)GG * HH, dwih1b, (size_t)GG * HH);
    conv(dWhh, dwhh0b, (size_t)GG * HH);
    conv(dWhh + (size_t)GG * HH, dwhh1b, (size_t)GG * HH);
    conv(doW, doWb, (size_t)DD * HH);
    k_decoder<<<NBLK, 256, 0, stream>>>(dwih0b, dwhh0b, dwih1b, dwhh1b, doWb,
        dbih, dbhh, dob, zbuf, c0l, c1l, h0d, h1d, xt, dec_out, bar);
}
